// AxialAttentionModule_523986010401
// MI455X (gfx1250) — compile-verified, matrix operand order checked by reading, not run on hardware
//
#include <hip/hip_runtime.h>

#define DEV __device__ __forceinline__

typedef __attribute__((ext_vector_type(16))) __bf16 v16bf;
typedef __attribute__((ext_vector_type(8)))  float v8f;

union FragU { uint4 u[2]; v16bf v; };

DEV unsigned short f2bf(float f) {
  unsigned u = __float_as_uint(f);
  return (unsigned short)((u + 0x7FFFu + ((u >> 16) & 1u)) >> 16);
}

// Load a 16-lane-striped bf16 WMMA fragment (A from M x K row-major rows, or
// B from N x K row-major rows). rowptr = lane's row base + k-chunk offset.
// Lane needs halves K = (L/16)*8 + {0..7} and +16..+23  -> two aligned b128 loads.
DEV v16bf load_frag(const unsigned short* rowptr, int lane) {
  const uint4* p = (const uint4*)(rowptr + ((lane >> 4) << 3));
  FragU r;
  r.u[0] = p[0];
  r.u[1] = p[2];
  return r.v;
}

DEV v8f wmma_bf16(v16bf a, v16bf b, v8f c) {
  return __builtin_amdgcn_wmma_f32_16x16x32_bf16(false, a, false, b, (short)0, c,
                                                 false, false);
}

// Row-start element offset into src-layout (N,D,B,E) for logical GEMM row gr.
// mode 0 (row attn): gr = s*128 + bt   -> contiguous.
// mode 1 (col attn): gr = dd*2048 + nb -> gather (n,dd,b).
DEV int xoff(int gr, int mode) {
  if (mode == 0) return gr << 8;
  int dd = gr >> 11, nb = gr & 2047;
  return (((nb >> 2) * 128) + dd * 4 + (nb & 3)) << 8;
}

__global__ void k_convert(const float* __restrict__ in, unsigned short* __restrict__ out, int n) {
  int i = blockIdx.x * 256 + threadIdx.x;
  if (i < n) out[i] = f2bf(in[i]);
}

// QKV projection: y(gr, n) = x(gr,:) . W(n,:) + b(n), n in [0,768)
// Writes Q,K as bf16 [bt][h][s][hd]; V transposed [bt][h][hd][s].
__global__ void k_qkv(const unsigned short* __restrict__ xbf,
                      const unsigned short* __restrict__ wbf,
                      const float* __restrict__ bias,
                      unsigned short* __restrict__ qb, unsigned short* __restrict__ kb,
                      unsigned short* __restrict__ vb,
                      int S, int btShift, int mode) {
  int lane = threadIdx.x & 31;
  int wv   = threadIdx.x >> 5;
  int mtile = blockIdx.x * 4 + wv;
  int n0b   = blockIdx.y * 64;
  int Bt    = 1 << btShift;

  const unsigned short* aptr = xbf + xoff(mtile * 16 + (lane & 15), mode);
  v8f c0 = {0,0,0,0,0,0,0,0}, c1 = c0, c2 = c0, c3 = c0;
#pragma unroll
  for (int kc = 0; kc < 256; kc += 32) {
    v16bf a = load_frag(aptr + kc, lane);
    c0 = wmma_bf16(a, load_frag(wbf + (n0b +  0 + (lane & 15)) * 256 + kc, lane), c0);
    c1 = wmma_bf16(a, load_frag(wbf + (n0b + 16 + (lane & 15)) * 256 + kc, lane), c1);
    c2 = wmma_bf16(a, load_frag(wbf + (n0b + 32 + (lane & 15)) * 256 + kc, lane), c2);
    c3 = wmma_bf16(a, load_frag(wbf + (n0b + 48 + (lane & 15)) * 256 + kc, lane), c3);
  }
  v8f cs[4] = {c0, c1, c2, c3};
#pragma unroll
  for (int j = 0; j < 4; j++) {
    int n = n0b + j * 16 + (lane & 15);
    float bv = bias[n];
    int which = n >> 8, e = n & 255, h = e >> 6, hd = e & 63;
#pragma unroll
    for (int g = 0; g < 8; g++) {
      int gr = mtile * 16 + g + ((lane >> 4) << 3);
      int s = gr >> btShift, bt = gr & (Bt - 1);
      unsigned short val = f2bf(cs[j][g] + bv);
      int bh = bt * 4 + h;
      if (which == 0)      qb[(bh * S + s) * 64 + hd] = val;
      else if (which == 1) kb[(bh * S + s) * 64 + hd] = val;
      else                 vb[(bh * 64 + hd) * S + s] = val;
    }
  }
}

// One wave per (q-tile of 16, head, batch): flash attention, online softmax.
__global__ void k_attn(const unsigned short* __restrict__ qb,
                       const unsigned short* __restrict__ kb,
                       const unsigned short* __restrict__ vb,
                       unsigned short* __restrict__ ob,
                       int S, int btShift, const int* __restrict__ sepp, int useMask) {
  __shared__ unsigned short Plds[16 * 32];
  int lane = threadIdx.x;
  int qt = blockIdx.x, h = blockIdx.y, bt = blockIdx.z;
  int Bt = 1 << btShift;
  int bh = bt * 4 + h;
  const unsigned short* qp = qb + (size_t)bh * S * 64;
  const unsigned short* kp = kb + (size_t)bh * S * 64;
  const unsigned short* vp = vb + (size_t)bh * 64 * S;
  int kvEnd = S;
  if (useMask) { int sep = sepp[0]; if (qt * 16 >= sep) kvEnd = sep; }

  v16bf qA0 = load_frag(qp + (qt * 16 + (lane & 15)) * 64, lane);
  v16bf qA1 = load_frag(qp + (qt * 16 + (lane & 15)) * 64 + 32, lane);

  const v8f vz = {0,0,0,0,0,0,0,0};
  v8f O0 = vz, O1 = vz, O2 = vz, O3 = vz;
  float mg[8], lg[8];
#pragma unroll
  for (int g = 0; g < 8; g++) { mg[g] = -1e30f; lg[g] = 0.0f; }
  const float scale = 0.125f;  // HD=64 ^ -0.5

  for (int kbase = 0; kbase < kvEnd; kbase += 32) {
    v8f s0v = vz, s1v = vz;
    const unsigned short* kr0 = kp + (kbase + (lane & 15)) * 64;
    const unsigned short* kr1 = kp + (kbase + 16 + (lane & 15)) * 64;
    s0v = wmma_bf16(qA0, load_frag(kr0, lane), s0v);
    s0v = wmma_bf16(qA1, load_frag(kr0 + 32, lane), s0v);
    s1v = wmma_bf16(qA0, load_frag(kr1, lane), s1v);
    s1v = wmma_bf16(qA1, load_frag(kr1 + 32, lane), s1v);

    float fO[8];
#pragma unroll
    for (int g = 0; g < 8; g++) {
      float s0 = s0v[g] * scale, s1 = s1v[g] * scale;
      float mn = fmaxf(s0, s1);
      mn = fmaxf(mn, __shfl_xor(mn, 1));
      mn = fmaxf(mn, __shfl_xor(mn, 2));
      mn = fmaxf(mn, __shfl_xor(mn, 4));
      mn = fmaxf(mn, __shfl_xor(mn, 8));   // 16-lane row reduction
      mn = fmaxf(mn, mg[g]);
      float f = __expf(mg[g] - mn);
      mg[g] = mn;
      float p0 = __expf(s0 - mn), p1 = __expf(s1 - mn);
      float rs = p0 + p1;
      rs += __shfl_xor(rs, 1);
      rs += __shfl_xor(rs, 2);
      rs += __shfl_xor(rs, 4);
      rs += __shfl_xor(rs, 8);
      lg[g] = lg[g] * f + rs;
      fO[g] = f;
      int row = g + ((lane >> 4) << 3);
      Plds[row * 32 + (lane & 15)]      = f2bf(p0);
      Plds[row * 32 + (lane & 15) + 16] = f2bf(p1);
    }
#pragma unroll
    for (int g = 0; g < 8; g++) {
      O0[g] *= fO[g]; O1[g] *= fO[g]; O2[g] *= fO[g]; O3[g] *= fO[g];
    }
    __syncthreads();
    v16bf pA = load_frag(Plds + (lane & 15) * 32, lane);
    O0 = wmma_bf16(pA, load_frag(vp + ( 0 + (lane & 15)) * S + kbase, lane), O0);
    O1 = wmma_bf16(pA, load_frag(vp + (16 + (lane & 15)) * S + kbase, lane), O1);
    O2 = wmma_bf16(pA, load_frag(vp + (32 + (lane & 15)) * S + kbase, lane), O2);
    O3 = wmma_bf16(pA, load_frag(vp + (48 + (lane & 15)) * S + kbase, lane), O3);
    __syncthreads();
  }
#pragma unroll
  for (int g = 0; g < 8; g++) {
    float inv = 1.0f / lg[g];
    int s  = qt * 16 + g + ((lane >> 4) << 3);
    int gr = s * Bt + bt;
    int eb = h * 64 + (lane & 15);
    ob[gr * 256 + eb +  0] = f2bf(O0[g] * inv);
    ob[gr * 256 + eb + 16] = f2bf(O1[g] * inv);
    ob[gr * 256 + eb + 32] = f2bf(O2[g] * inv);
    ob[gr * 256 + eb + 48] = f2bf(O3[g] * inv);
  }
}

// Out-projection: mode 0: acc = src + O.W^T + b ; mode 1: acc += O.W^T + b
__global__ void k_outproj(const unsigned short* __restrict__ ob,
                          const unsigned short* __restrict__ wbf,
                          const float* __restrict__ bias,
                          const float* __restrict__ src,
                          float* __restrict__ acc, int mode) {
  int lane = threadIdx.x & 31;
  int wv   = threadIdx.x >> 5;
  int mtile = blockIdx.x * 4 + wv;
  int n0b   = blockIdx.y * 64;
  const unsigned short* aptr = ob + (mtile * 16 + (lane & 15)) * 256;
  v8f c0 = {0,0,0,0,0,0,0,0}, c1 = c0, c2 = c0, c3 = c0;
#pragma unroll
  for (int kc = 0; kc < 256; kc += 32) {
    v16bf a = load_frag(aptr + kc, lane);
    c0 = wmma_bf16(a, load_frag(wbf + (n0b +  0 + (lane & 15)) * 256 + kc, lane), c0);
    c1 = wmma_bf16(a, load_frag(wbf + (n0b + 16 + (lane & 15)) * 256 + kc, lane), c1);
    c2 = wmma_bf16(a, load_frag(wbf + (n0b + 32 + (lane & 15)) * 256 + kc, lane), c2);
    c3 = wmma_bf16(a, load_frag(wbf + (n0b + 48 + (lane & 15)) * 256 + kc, lane), c3);
  }
  v8f cs[4] = {c0, c1, c2, c3};
#pragma unroll
  for (int j = 0; j < 4; j++) {
    int n = n0b + j * 16 + (lane & 15);
    float bv = bias[n];
#pragma unroll
    for (int g = 0; g < 8; g++) {
      int gr = mtile * 16 + g + ((lane >> 4) << 3);
      int off = xoff(gr, mode) + n;
      float val = cs[j][g] + bv;
      if (mode == 0) acc[off] = src[off] + val;
      else           acc[off] += val;
    }
  }
}

// LayerNorm over E=256: one wave per row, 8 elems/lane.
__global__ void k_ln(const float* __restrict__ acc, const float* __restrict__ g,
                     const float* __restrict__ b, float* __restrict__ out) {
  int lane = threadIdx.x & 31;
  int wv   = threadIdx.x >> 5;
  int row  = blockIdx.x * 8 + wv;
  const float* x = acc + row * 256;
  float v[8], s = 0.0f, sq = 0.0f;
#pragma unroll
  for (int i = 0; i < 8; i++) {
    v[i] = x[lane + i * 32];
    s += v[i];
    sq += v[i] * v[i];
  }
#pragma unroll
  for (int m = 1; m < 32; m <<= 1) { s += __shfl_xor(s, m); sq += __shfl_xor(sq, m); }
  float mu  = s * (1.0f / 256.0f);
  float var = sq * (1.0f / 256.0f) - mu * mu;
  float inv = rsqrtf(var + 1e-5f);
#pragma unroll
  for (int i = 0; i < 8; i++) {
    int e = lane + i * 32;
    out[row * 256 + e] = (v[i] - mu) * inv * g[e] + b[e];
  }
}

extern "C" void kernel_launch(void* const* d_in, const int* in_sizes, int n_in,
                              void* d_out, int out_size, void* d_ws, size_t ws_size,
                              hipStream_t stream) {
  (void)in_sizes; (void)n_in; (void)out_size; (void)ws_size;
  const float* src       = (const float*)d_in[0];
  const float* w_in_row  = (const float*)d_in[1];
  const float* b_in_row  = (const float*)d_in[2];
  const float* w_out_row = (const float*)d_in[3];
  const float* b_out_row = (const float*)d_in[4];
  const float* w_in_col  = (const float*)d_in[5];
  const float* b_in_col  = (const float*)d_in[6];
  const float* w_out_col = (const float*)d_in[7];
  const float* b_out_col = (const float*)d_in[8];
  const float* ln_g      = (const float*)d_in[9];
  const float* ln_b      = (const float*)d_in[10];
  const int*   sep       = (const int*)d_in[12];
  float* out = (float*)d_out;

  const int NE = 512 * 32 * 4 * 256;  // 16,777,216
  char* ws = (char*)d_ws;
  size_t off = 0;
  auto alloc = [&](size_t bytes) { void* p = ws + off; off += (bytes + 255) & ~(size_t)255; return p; };
  unsigned short* xbf = (unsigned short*)alloc((size_t)NE * 2);
  unsigned short* wir = (unsigned short*)alloc(768 * 256 * 2);
  unsigned short* wor = (unsigned short*)alloc(256 * 256 * 2);
  unsigned short* wic = (unsigned short*)alloc(768 * 256 * 2);
  unsigned short* woc = (unsigned short*)alloc(256 * 256 * 2);
  unsigned short* qb  = (unsigned short*)alloc((size_t)NE * 2);
  unsigned short* kb  = (unsigned short*)alloc((size_t)NE * 2);
  unsigned short* vb  = (unsigned short*)alloc((size_t)NE * 2);
  unsigned short* ob  = (unsigned short*)alloc((size_t)NE * 2);
  float*          acc = (float*)alloc((size_t)NE * 4);

  k_convert<<<NE / 256, 256, 0, stream>>>(src, xbf, NE);
  k_convert<<<768, 256, 0, stream>>>(w_in_row, wir, 768 * 256);
  k_convert<<<256, 256, 0, stream>>>(w_out_row, wor, 256 * 256);
  k_convert<<<768, 256, 0, stream>>>(w_in_col, wic, 768 * 256);
  k_convert<<<256, 256, 0, stream>>>(w_out_col, woc, 256 * 256);

  dim3 gq(1024, 12), go(1024, 4);
  // Row attention: S=512, Bt=128 (shift 7), masked
  k_qkv<<<gq, 128, 0, stream>>>(xbf, wir, b_in_row, qb, kb, vb, 512, 7, 0);
  dim3 gar(32, 4, 128);
  k_attn<<<gar, 32, 0, stream>>>(qb, kb, vb, ob, 512, 7, sep, 1);
  k_outproj<<<go, 128, 0, stream>>>(ob, wor, b_out_row, src, acc, 0);
  // Column attention: S=32, Bt=2048 (shift 11), no mask
  k_qkv<<<gq, 128, 0, stream>>>(xbf, wic, b_in_col, qb, kb, vb, 32, 11, 1);
  dim3 gac(2, 4, 2048);
  k_attn<<<gac, 32, 0, stream>>>(qb, kb, vb, ob, 32, 11, sep, 0);
  k_outproj<<<go, 128, 0, stream>>>(ob, woc, b_out_col, src, acc, 1);

  k_ln<<<8192, 256, 0, stream>>>(acc, ln_g, ln_b, out);
}
